// MoTLlamaAttention_6219112644803
// MI455X (gfx1250) — matrix operand order repaired and checked by reading, not compiled, not run on hardware
//
#include <hip/hip_runtime.h>
#include <hip/hip_bf16.h>
#include <math.h>

typedef __bf16 bf16_t;
typedef __attribute__((ext_vector_type(8)))  __bf16 bf16x8_t;
typedef __attribute__((ext_vector_type(16))) __bf16 bf16x16_t;
typedef __attribute__((ext_vector_type(8)))  float  f32x8_t;

#define Bsz 2
#define Sq 1024
#define Dm 2048
#define NHh 32
#define NKVh 8
#define HDd 64
#define NTOK (Bsz * Sq)              // 2048 tokens
#define K2D  (2 * Dm)                // 4096 (concat-modality K)
#define QKVC (NHh*HDd + 2*NKVh*HDd)  // 3072 fused QKV cols

__device__ __forceinline__ f32x8_t wmma_bf16(bf16x16_t a, bf16x16_t b, f32x8_t c) {
  return __builtin_amdgcn_wmma_f32_16x16x32_bf16(false, a, false, b, (short)0, c, false, false);
}

__device__ __forceinline__ bf16x16_t cat8(bf16x8_t lo, bf16x8_t hi) {
  return __builtin_shufflevector(lo, hi, 0,1,2,3,4,5,6,7,8,9,10,11,12,13,14,15);
}

// ---------------------------------------------------------------------------
// Phase 0: build bf16 operands (masked concat hidden, fused QKV weight, Wo)
// ---------------------------------------------------------------------------
__global__ void build_masked_hidden(const float* __restrict__ h0, const float* __restrict__ h1,
                                    const int* __restrict__ v0, const int* __restrict__ v1,
                                    bf16_t* __restrict__ A) {
  int idx = blockIdx.x * blockDim.x + threadIdx.x;     // over NTOK*K2D
  int t = idx >> 12, c = idx & 4095;
  float val = (c < Dm) ? (v0[t] ? h0[(size_t)t * Dm + c] : 0.0f)
                       : (v1[t] ? h1[(size_t)t * Dm + (c - Dm)] : 0.0f);
  A[idx] = (bf16_t)val;
}

__global__ void build_wqkv(const float* __restrict__ Wq0, const float* __restrict__ Wq1,
                           const float* __restrict__ Wk0, const float* __restrict__ Wk1,
                           const float* __restrict__ Wv0, const float* __restrict__ Wv1,
                           bf16_t* __restrict__ W) {
  int idx = blockIdx.x * blockDim.x + threadIdx.x;     // over QKVC*K2D
  int r = idx >> 12, c = idx & 4095;
  int mod = c >> 11, cc = c & 2047;
  const float* src; int row;
  if (r < 2048)      { src = mod ? Wq1 : Wq0; row = r; }
  else if (r < 2560) { src = mod ? Wk1 : Wk0; row = r - 2048; }
  else               { src = mod ? Wv1 : Wv0; row = r - 2560; }
  W[idx] = (bf16_t)src[(size_t)row * Dm + cc];
}

__global__ void build_wo(const float* __restrict__ Wo0, const float* __restrict__ Wo1,
                         bf16_t* __restrict__ W) {
  int idx = blockIdx.x * blockDim.x + threadIdx.x;     // over (2*Dm)*Dm
  int r = idx >> 11, c = idx & 2047;
  W[idx] = (bf16_t)((r < Dm) ? Wo0[(size_t)r * Dm + c] : Wo1[(size_t)(r - Dm) * Dm + c]);
}

// ---------------------------------------------------------------------------
// NT GEMM: C[M,N] = A[M,K] * B[N,K]^T, bf16 in / f32 acc out.
// Block tile 256x128, 8 waves (4M x 2N), wave tile 64x64 -> 16 WMMA per
// k-slab with 1:1 ds_load:wmma ratio. Double-buffered LDS filled with
// GLOBAL_LOAD_ASYNC_TO_LDS_B128 (ASYNCcnt); last iteration peeled so the
// steady-state loop is branch-free.
// mask_mode==1: fused dual-modality masked writeback for the output proj.
// ---------------------------------------------------------------------------
__global__ __launch_bounds__(256)
void gemm_nt_bf16(const bf16_t* __restrict__ A, const bf16_t* __restrict__ Bm,
                  float* __restrict__ C, int M, int N, int K,
                  int mask_mode, const int* __restrict__ v0, const int* __restrict__ v1,
                  float* __restrict__ out0, float* __restrict__ out1) {
  __shared__ alignas(32) bf16_t sA[2][256][32];   // 32 KB
  __shared__ alignas(32) bf16_t sB[2][128][32];   // 16 KB
  const int t = threadIdx.x;
  const int lane = t & 31, w = t >> 5;
  const int half = lane >> 4, l15 = lane & 15;
  const int wm = w >> 1, wn = w & 1;
  const int m0 = blockIdx.y * 256, n0 = blockIdx.x * 128;

  // Async staging: A tile 16KB -> 4x16B per thread, B tile 8KB -> 2x16B.
  const int r0 = t >> 2;               // rows 0..63 (chunks at +64/+128/+192)
  const int c8 = (t & 3) * 8;          // bf16 col within 32-wide k-slab
  const unsigned baseA = (unsigned)(unsigned long long)(const void*)&sA[0][0][0];
  const unsigned baseB = (unsigned)(unsigned long long)(const void*)&sB[0][0][0];
  const unsigned offA = baseA + (unsigned)((r0 * 32 + c8) * 2);   // +4096 per 64 rows
  const unsigned offB = baseB + (unsigned)((r0 * 32 + c8) * 2);
  const bf16_t* gA = A  + (size_t)(m0 + r0) * K + c8;
  const bf16_t* gB = Bm + (size_t)(n0 + r0) * K + c8;
  const size_t rstepA = (size_t)64 * K;           // +64 rows in elements

  f32x8_t acc[4][4];
  const f32x8_t zf = {0.f,0.f,0.f,0.f,0.f,0.f,0.f,0.f};
#pragma unroll
  for (int mi = 0; mi < 4; ++mi)
#pragma unroll
    for (int ni = 0; ni < 4; ++ni) acc[mi][ni] = zf;

  const int nk = K >> 5;

#define STAGE_TILE(ki)                                                         \
  do {                                                                         \
    const unsigned boA = ((ki) & 1) ? 16384u : 0u;                             \
    const unsigned boB = ((ki) & 1) ? 8192u  : 0u;                             \
    const size_t ko = (size_t)(ki) << 5;                                       \
    asm volatile("global_load_async_to_lds_b128 %0, %1, off"                   \
                 :: "v"(offA + boA), "v"(gA + ko) : "memory");                 \
    asm volatile("global_load_async_to_lds_b128 %0, %1, off"                   \
                 :: "v"(offA + boA + 4096u),  "v"(gA + rstepA + ko) : "memory");\
    asm volatile("global_load_async_to_lds_b128 %0, %1, off"                   \
                 :: "v"(offA + boA + 8192u),  "v"(gA + 2*rstepA + ko) : "memory");\
    asm volatile("global_load_async_to_lds_b128 %0, %1, off"                   \
                 :: "v"(offA + boA + 12288u), "v"(gA + 3*rstepA + ko) : "memory");\
    asm volatile("global_load_async_to_lds_b128 %0, %1, off"                   \
                 :: "v"(offB + boB), "v"(gB + ko) : "memory");                 \
    asm volatile("global_load_async_to_lds_b128 %0, %1, off"                   \
                 :: "v"(offB + boB + 4096u), "v"(gB + rstepA /*==64*K*/ + ko) : "memory");\
  } while (0)

  auto compute_slab = [&](int cur) {
    bf16x16_t a[4];
#pragma unroll
    for (int mi = 0; mi < 4; ++mi) {
      const int r = wm * 64 + mi * 16 + l15;
      a[mi] = cat8(*(const bf16x8_t*)&sA[cur][r][half * 16],
                   *(const bf16x8_t*)&sA[cur][r][half * 16 + 8]);
    }
#pragma unroll
    for (int ni = 0; ni < 4; ++ni) {
      bf16x16_t b = *(const bf16x16_t*)&sB[cur][wn * 64 + ni * 16 + l15][half * 16];
#pragma unroll
      for (int mi = 0; mi < 4; ++mi) acc[mi][ni] = wmma_bf16(a[mi], b, acc[mi][ni]);
    }
  };

  STAGE_TILE(0);
  for (int ki = 0; ki < nk - 1; ++ki) {
    STAGE_TILE(ki + 1);
    // async loads complete in order: <=6 outstanding means tile ki landed
    asm volatile("s_wait_asynccnt 6" ::: "memory");
    __syncthreads();
    compute_slab(ki & 1);
    __syncthreads();   // all waves done with buf before it is re-staged
  }
  asm volatile("s_wait_asynccnt 0" ::: "memory");
  __syncthreads();
  compute_slab((nk - 1) & 1);
#undef STAGE_TILE

#pragma unroll
  for (int mi = 0; mi < 4; ++mi)
#pragma unroll
    for (int ni = 0; ni < 4; ++ni) {
      const int col = n0 + wn * 64 + ni * 16 + l15;
#pragma unroll
      for (int r = 0; r < 8; ++r) {
        const int row = m0 + wm * 64 + mi * 16 + r + 8 * half;
        const float v = acc[mi][ni][r];
        if (mask_mode == 0) {
          C[(size_t)row * N + col] = v;
        } else {
          if (col < Dm) out0[(size_t)row * Dm + col]        = v0[row] ? v : 0.0f;
          else          out1[(size_t)row * Dm + (col - Dm)] = v1[row] ? v : 0.0f;
        }
      }
    }
}

// ---------------------------------------------------------------------------
// RoPE + layout: qkv f32 [NTOK,3072] -> q bf16 [B,NH,S,HD], k bf16 [B,NKV,S,HD],
// v^T bf16 [B,NKV,HD,S] (transposed so P*V is an NT GEMM too).
// ---------------------------------------------------------------------------
__global__ void rope_scatter(const float* __restrict__ qkv, const int* __restrict__ pos,
                             bf16_t* __restrict__ qb, bf16_t* __restrict__ kb,
                             bf16_t* __restrict__ vtb) {
  const int idx = blockIdx.x * blockDim.x + threadIdx.x;   // over NTOK*QKVC
  const int tok = idx / QKVC;
  const int c = idx - tok * QKVC;
  const int b = tok >> 10, s = tok & 1023;
  const float x = qkv[idx];
  if (c < NHh * HDd) {                       // Q : rope
    const int h = c >> 6, i = c & 63;
    const float ang = (float)pos[tok] * expf(-((float)(i & 31) / 32.0f) * logf(10000.0f));
    const float pr = qkv[idx ^ 32];          // rotate_half partner (bit5 of i)
    const float o = x * cosf(ang) + ((i < 32) ? -pr : pr) * sinf(ang);
    qb[(((size_t)(b * NHh + h) * Sq + s) << 6) + i] = (bf16_t)o;
  } else if (c < NHh * HDd + NKVh * HDd) {   // K : rope
    const int c2 = c - NHh * HDd;
    const int h = c2 >> 6, i = c2 & 63;
    const float ang = (float)pos[tok] * expf(-((float)(i & 31) / 32.0f) * logf(10000.0f));
    const float pr = qkv[idx ^ 32];
    const float o = x * cosf(ang) + ((i < 32) ? -pr : pr) * sinf(ang);
    kb[(((size_t)(b * NKVh + h) * Sq + s) << 6) + i] = (bf16_t)o;
  } else {                                   // V : store transposed
    const int c2 = c - (NHh * HDd + NKVh * HDd);
    const int h = c2 >> 6, i = c2 & 63;
    vtb[((size_t)(b * NKVh + h) * HDd + i) * Sq + s] = (bf16_t)x;
  }
}

// ---------------------------------------------------------------------------
// Causal flash attention. grid (S/64, B*NH), 128 thr = 4 waves, 16 q-rows/wave.
// Row softmax stays inside 16-lane halves (wave32 C-fragment layout).
// ---------------------------------------------------------------------------
__global__ __launch_bounds__(128)
void flash_attn(const bf16_t* __restrict__ qb, const bf16_t* __restrict__ kb,
                const bf16_t* __restrict__ vtb, bf16_t* __restrict__ ctx) {
  __shared__ alignas(32) bf16_t sQ[64][64];
  __shared__ alignas(32) bf16_t sK[64][64];
  __shared__ alignas(32) bf16_t sVT[64][64];
  __shared__ alignas(32) bf16_t sP[4][16][64];

  const int t = threadIdx.x;
  const int lane = t & 31, w = t >> 5;
  const int half = lane >> 4, l15 = lane & 15;
  const int qt = blockIdx.x, q0 = qt * 64;
  const int bh = blockIdx.y;
  const int b = bh / NHh, h = bh - b * NHh;
  const int kvh = h / (NHh / NKVh);

  const bf16_t* qbase = qb  + (size_t)(b * NHh + h)   * Sq * HDd;
  const bf16_t* kbas  = kb  + (size_t)(b * NKVh + kvh) * Sq * HDd;
  const bf16_t* vtbas = vtb + (size_t)(b * NKVh + kvh) * HDd * Sq;

#pragma unroll
  for (int it = 0; it < 2; ++it) {
    int seg = t + it * 128, r = seg >> 2, cc = (seg & 3) * 16;
    *(bf16x16_t*)&sQ[r][cc] = *(const bf16x16_t*)&qbase[(size_t)(q0 + r) * HDd + cc];
  }
  __syncthreads();

  bf16x16_t aq[2];
#pragma unroll
  for (int kbi = 0; kbi < 2; ++kbi) {
    const int kb2 = kbi * 32, r = w * 16 + l15;
    aq[kbi] = cat8(*(const bf16x8_t*)&sQ[r][kb2 + half * 16],
                   *(const bf16x8_t*)&sQ[r][kb2 + half * 16 + 8]);
  }

  const f32x8_t zf = {0.f,0.f,0.f,0.f,0.f,0.f,0.f,0.f};
  f32x8_t oacc[4];
#pragma unroll
  for (int hf = 0; hf < 4; ++hf) oacc[hf] = zf;
  float mi_[8], li_[8];
#pragma unroll
  for (int r = 0; r < 8; ++r) { mi_[r] = -3.0e38f; li_[r] = 0.0f; }

  for (int kt = 0; kt <= qt; ++kt) {
    const int kb0 = kt * 64;
    __syncthreads();
#pragma unroll
    for (int it = 0; it < 2; ++it) {
      int seg = t + it * 128, r = seg >> 2, cc = (seg & 3) * 16;
      *(bf16x16_t*)&sK[r][cc]  = *(const bf16x16_t*)&kbas [(size_t)(kb0 + r) * HDd + cc];
      *(bf16x16_t*)&sVT[r][cc] = *(const bf16x16_t*)&vtbas[(size_t)r * Sq + kb0 + cc];
    }
    __syncthreads();

    f32x8_t sc[4];
#pragma unroll
    for (int nf = 0; nf < 4; ++nf) sc[nf] = zf;
#pragma unroll
    for (int nf = 0; nf < 4; ++nf)
#pragma unroll
      for (int kbi = 0; kbi < 2; ++kbi) {
        bf16x16_t bk = *(const bf16x16_t*)&sK[nf * 16 + l15][kbi * 32 + half * 16];
        sc[nf] = wmma_bf16(aq[kbi], bk, sc[nf]);
      }

    float ps[4][8];
    const bool diag = (kt == qt);
#pragma unroll
    for (int nf = 0; nf < 4; ++nf)
#pragma unroll
      for (int r = 0; r < 8; ++r) {
        float v = sc[nf][r] * 0.125f;                 // 1/sqrt(64)
        if (diag) {
          const int col = kb0 + nf * 16 + l15;
          const int row = q0 + w * 16 + r + 8 * half;
          if (col > row) v = -3.0e38f;
        }
        ps[nf][r] = v;
      }

#pragma unroll
    for (int r = 0; r < 8; ++r) {
      float mx = fmaxf(fmaxf(ps[0][r], ps[1][r]), fmaxf(ps[2][r], ps[3][r]));
#pragma unroll
      for (int mm = 1; mm < 16; mm <<= 1) mx = fmaxf(mx, __shfl_xor(mx, mm, 32));
      const float mnew = fmaxf(mi_[r], mx);
      const float corr = exp2f((mi_[r] - mnew) * 1.44269504f);
      mi_[r] = mnew;
      li_[r] *= corr;
#pragma unroll
      for (int hf = 0; hf < 4; ++hf) oacc[hf][r] *= corr;
      float sum = 0.0f;
#pragma unroll
      for (int nf = 0; nf < 4; ++nf) {
        const float p = exp2f((ps[nf][r] - mnew) * 1.44269504f);
        sP[w][r + 8 * half][nf * 16 + l15] = (bf16_t)p;
        sum += p;
      }
#pragma unroll
      for (int mm = 1; mm < 16; mm <<= 1) sum += __shfl_xor(sum, mm, 32);
      li_[r] += sum;
    }

#pragma unroll
    for (int kbi = 0; kbi < 2; ++kbi) {
      const int kb2 = kbi * 32;
      bf16x16_t ap = cat8(*(const bf16x8_t*)&sP[w][l15][kb2 + half * 16],
                          *(const bf16x8_t*)&sP[w][l15][kb2 + half * 16 + 8]);
#pragma unroll
      for (int hf = 0; hf < 4; ++hf) {
        bf16x16_t bv = *(const bf16x16_t*)&sVT[hf * 16 + l15][kb2 + half * 16];
        oacc[hf] = wmma_bf16(ap, bv, oacc[hf]);
      }
    }
  }

#pragma unroll
  for (int hf = 0; hf < 4; ++hf)
#pragma unroll
    for (int r = 0; r < 8; ++r) {
      const int srow = q0 + w * 16 + r + 8 * half;
      const float v = oacc[hf][r] / li_[r];
      ctx[(size_t)(b * Sq + srow) * (NHh * HDd) + h * HDd + hf * 16 + l15] = (bf16_t)v;
    }
}

// ---------------------------------------------------------------------------
extern "C" void kernel_launch(void* const* d_in, const int* in_sizes, int n_in,
                              void* d_out, int out_size, void* d_ws, size_t ws_size,
                              hipStream_t stream) {
  const float* h0  = (const float*)d_in[0];
  const float* h1  = (const float*)d_in[1];
  const int*   v0  = (const int*)d_in[2];
  const int*   v1  = (const int*)d_in[3];
  const int*   pos = (const int*)d_in[4];
  const float* Wq0 = (const float*)d_in[5];
  const float* Wq1 = (const float*)d_in[6];
  const float* Wk0 = (const float*)d_in[7];
  const float* Wk1 = (const float*)d_in[8];
  const float* Wv0 = (const float*)d_in[9];
  const float* Wv1 = (const float*)d_in[10];
  const float* Wo0 = (const float*)d_in[11];
  const float* Wo1 = (const float*)d_in[12];

  char* ws = (char*)d_ws;
  size_t off = 0;
  auto alloc = [&](size_t bytes) {
    void* p = ws + off;
    off += (bytes + 255) & ~(size_t)255;
    return p;
  };
  bf16_t* Amask = (bf16_t*)alloc((size_t)NTOK * K2D * 2);            // 16 MB
  bf16_t* Wqkv  = (bf16_t*)alloc((size_t)QKVC * K2D * 2);            // 24 MB
  float*  qkv   = (float*) alloc((size_t)NTOK * QKVC * 4);           // 24 MB
  bf16_t* qb    = (bf16_t*)alloc((size_t)Bsz * NHh * Sq * HDd * 2);  // 8 MB
  bf16_t* kbuf  = (bf16_t*)alloc((size_t)Bsz * NKVh * Sq * HDd * 2); // 2 MB
  bf16_t* vtb   = (bf16_t*)alloc((size_t)Bsz * NKVh * HDd * Sq * 2); // 2 MB
  bf16_t* ctxb  = (bf16_t*)alloc((size_t)NTOK * NHh * HDd * 2);      // 8 MB
  bf16_t* Wob   = (bf16_t*)alloc((size_t)(2 * Dm) * Dm * 2);         // 16 MB

  float* out0 = (float*)d_out;
  float* out1 = out0 + (size_t)NTOK * Dm;

  build_masked_hidden<<<(NTOK * K2D) / 256, 256, 0, stream>>>(h0, h1, v0, v1, Amask);
  build_wqkv<<<(QKVC * K2D) / 256, 256, 0, stream>>>(Wq0, Wq1, Wk0, Wk1, Wv0, Wv1, Wqkv);
  build_wo<<<((2 * Dm) * Dm) / 256, 256, 0, stream>>>(Wo0, Wo1, Wob);

  gemm_nt_bf16<<<dim3(QKVC / 128, NTOK / 256), 256, 0, stream>>>(
      Amask, Wqkv, qkv, NTOK, QKVC, K2D, 0, nullptr, nullptr, nullptr, nullptr);

  rope_scatter<<<(NTOK * QKVC) / 256, 256, 0, stream>>>(qkv, pos, qb, kbuf, vtb);

  flash_attn<<<dim3(Sq / 64, Bsz * NHh), 128, 0, stream>>>(qb, kbuf, vtb, ctxb);

  gemm_nt_bf16<<<dim3((2 * Dm) / 128, NTOK / 256), 256, 0, stream>>>(
      ctxb, Wob, nullptr, NTOK, 2 * Dm, Dm, 1, v0, v1, out0, out1);
}
